// MHRetention_78640851189992
// MI455X (gfx1250) — matrix operand order repaired and checked by reading, not compiled, not run on hardware
//
#include <hip/hip_runtime.h>
#include <hip/hip_bf16.h>

#define DEV_INLINE __device__ __forceinline__

typedef __bf16 bf16;
typedef __attribute__((ext_vector_type(16))) __bf16 v16bf;
typedef __attribute__((ext_vector_type(8)))  __bf16 v8bf;
typedef __attribute__((ext_vector_type(8)))  float  v8f;

union Frag { v16bf v; v8bf h[2]; };

// Model dimensions (match reference setup_inputs()).
static constexpr int Bsz = 4;
static constexpr int Lsz = 4096;
static constexpr int Dsz = 2048;
static constexpr int Hn  = 16;
static constexpr int Dh  = 128;           // Dsz / Hn
static constexpr int Msz = Bsz * Lsz;     // 16384

static constexpr float INV_SCALE = 0.08838834764831845f; // 1/sqrt(128)

enum { ACT_RELU_SCALE = 0, ACT_SILU = 1, ACT_NONE_BF16 = 2, ACT_NONE_F32 = 3 };

// ---- CDNA5 async global->LDS DMA (ASYNCcnt-tracked, bypasses VGPRs) ----
// GV mode: vdst = LDS byte address (generic ptr low 32 bits == LDS offset per
// ISA flat-address truncation), vaddr = 64-bit global address pair.
DEV_INLINE void async_ld16(void* lds, const void* g) {
  asm volatile("global_load_async_to_lds_b128 %0, %1, off"
               :
               : "v"((unsigned)(uintptr_t)lds),
                 "v"((unsigned long long)(uintptr_t)g)
               : "memory");
}
DEV_INLINE void async_wait_all() {
  asm volatile("s_wait_asynccnt 0x0" ::: "memory");
}

// One K=32 step for an 8-wave workgroup computing a 128x128 tile.
// Wave (wm in 0..1, wn in 0..3) owns rows [wm*64,+64) x cols [wn*32,+32):
// 4x2 accumulators of v_wmma_f32_16x16x32_bf16.
//
// Fragment layouts per CDNA5 ISA (wave32):
//   A (16x32): lane -> M = lane&15 ; element j -> K = 8*(lane>>4) + (j&7) + (j>=8 ? 16 : 0)
//   B (32x16): lane -> N = lane&15 ; element j -> K = 8*(lane>>4) + (j&7) + (j>=8 ? 16 : 0)
//   C/D f32  : lane -> N = lane&15 ; vgpr r -> M = r + 8*(lane>>4)
template<int LDA, int LDB>
DEV_INLINE void wmma_tile_step(const bf16* sA, const bf16* sB,
                               int wm, int wn, int lane, v8f acc[4][2]) {
  const int kb  = (lane >> 4) * 8;
  const int am0 = wm * 64 + (lane & 15);
  const int bk  = (lane & 15) + ((lane >> 4) << 4);
  Frag a[4];
#pragma unroll
  for (int mi = 0; mi < 4; ++mi) {
    const bf16* p = sA + (am0 + mi * 16) * LDA + kb;
    a[mi].h[0] = *(const v8bf*)(p);        // K = kb .. kb+7
    a[mi].h[1] = *(const v8bf*)(p + 16);   // K = kb+16 .. kb+23
  }
  Frag bfr[2];
#pragma unroll
  for (int ni = 0; ni < 2; ++ni) {
    const bf16* p = sB + (lane & 15) + wn * 32 + ni * 16;
#pragma unroll
    for (int s = 0; s < 8; ++s) {
      bfr[ni].v[s]     = p[(kb + s) * LDB];
      bfr[ni].v[8 + s] = p[(kb + 16 + s) * LDB];
    }
  }
#pragma unroll
  for (int mi = 0; mi < 4; ++mi)
#pragma unroll
    for (int ni = 0; ni < 2; ++ni)
      acc[mi][ni] = __builtin_amdgcn_wmma_f32_16x16x32_bf16(
          false, a[mi].v, false, bfr[ni].v, (short)0, acc[mi][ni], false, false);
}

// C[M,N] = act(A[M,K] @ W[K,N]).  A is f32 or bf16; W is f32 (reference weights).
// 128x128 workgroup tile, BK=32, double-buffered LDS (one barrier per K-step).
// bf16 A input (final GEMM) is staged with async global->LDS DMA.
template<typename TIN, int ACT>
__global__ __launch_bounds__(256)
void gemm128_kernel(const TIN* __restrict__ A, const float* __restrict__ W,
                    void* __restrict__ outp, int Mdim, int Ndim, int Kdim) {
  __shared__ __align__(16) bf16 sA[2][128 * 32];   // [m][k]
  __shared__ __align__(16) bf16 sB[2][32 * 128];   // [k][n]
  const int tid  = threadIdx.x;
  const int lane = tid & 31, wave = tid >> 5;
  const int wm = wave >> 2, wn = wave & 3;
  const int gm0 = blockIdx.y * 128, gn0 = blockIdx.x * 128;

  const int ar = tid >> 1, ac0 = (tid & 1) * 16;   // A staging: row, col of 16
  const int br = tid >> 3, bc0 = (tid & 7) * 16;   // B staging: row, col of 16

  auto stage = [&](int buf, int k0) {
    const TIN* srcA = A + (size_t)(gm0 + ar) * Kdim + k0 + ac0;
    if constexpr (__is_same(TIN, bf16)) {
      async_ld16(&sA[buf][ar * 32 + ac0], srcA);        // 16B DMA
      async_ld16(&sA[buf][ar * 32 + ac0 + 8], srcA + 8);
    } else {
#pragma unroll
      for (int j = 0; j < 16; ++j) sA[buf][ar * 32 + ac0 + j] = (bf16)(float)srcA[j];
    }
    const float* srcB = W + (size_t)(k0 + br) * Ndim + gn0 + bc0;
#pragma unroll
    for (int j = 0; j < 16; ++j) sB[buf][br * 128 + bc0 + j] = (bf16)srcB[j];
  };

  v8f acc[4][2] = {};
  stage(0, 0);
  async_wait_all();
  __syncthreads();
  int cur = 0;
  for (int k0 = 0; k0 < Kdim; k0 += 32) {
    const int nxt = cur ^ 1;
    if (k0 + 32 < Kdim) stage(nxt, k0 + 32);            // overlaps with WMMA
    wmma_tile_step<32, 128>(sA[cur], sB[cur], wm, wn, lane, acc);
    async_wait_all();
    __syncthreads();
    cur = nxt;
  }

  const int hi = lane >> 4, ln = lane & 15;
#pragma unroll
  for (int mi = 0; mi < 4; ++mi)
#pragma unroll
    for (int ni = 0; ni < 2; ++ni)
#pragma unroll
      for (int r = 0; r < 8; ++r) {
        const int m = gm0 + wm * 64 + mi * 16 + hi * 8 + r;
        const int n = gn0 + wn * 32 + ni * 16 + ln;
        const float v = acc[mi][ni][r];
        const size_t idx = (size_t)m * Ndim + n;
        if (ACT == ACT_RELU_SCALE) {
          ((bf16*)outp)[idx] = (bf16)(fmaxf(v, 0.0f) * INV_SCALE);
        } else if (ACT == ACT_SILU) {
          ((bf16*)outp)[idx] = (bf16)(v / (1.0f + __expf(-v)));
        } else if (ACT == ACT_NONE_BF16) {
          ((bf16*)outp)[idx] = (bf16)v;
        } else {
          ((float*)outp)[idx] = v;
        }
      }
}

// kv[b,h] = K[b,h]^T @ V[b,h] : M=N=128 (dh), K=L=4096. One block per (b,h).
// V tile staged via async DMA; K tile staged transposed through VGPRs.
__global__ __launch_bounds__(256)
void kv_kernel(const bf16* __restrict__ Kb, const bf16* __restrict__ Vb,
               bf16* __restrict__ kvb) {
  __shared__ __align__(16) bf16 sKT[2][128 * 32];  // [dh][l]  (A, pre-transposed)
  __shared__ __align__(16) bf16 sV [2][32 * 128];  // [l][dh]  (B)
  const int bh = blockIdx.x;
  const int b = bh >> 4, h = bh & 15;
  const int tid  = threadIdx.x;
  const int lane = tid & 31, wave = tid >> 5;
  const int wm = wave >> 2, wn = wave & 3;
  const size_t rowbase = (size_t)b * Lsz;
  const int l = tid >> 3, c0 = (tid & 7) * 16;

  auto stage = [&](int buf, int l0) {
    const bf16* ks = Kb + (rowbase + l0 + l) * Dsz + h * Dh + c0;
    const bf16* vs = Vb + (rowbase + l0 + l) * Dsz + h * Dh + c0;
    async_ld16(&sV[buf][l * 128 + c0], vs);             // DMA, no VGPR round-trip
    async_ld16(&sV[buf][l * 128 + c0 + 8], vs + 8);
#pragma unroll
    for (int j = 0; j < 16; ++j) sKT[buf][(c0 + j) * 32 + l] = ks[j];  // transpose
  };

  v8f acc[4][2] = {};
  stage(0, 0);
  async_wait_all();
  __syncthreads();
  int cur = 0;
  for (int l0 = 0; l0 < Lsz; l0 += 32) {
    const int nxt = cur ^ 1;
    if (l0 + 32 < Lsz) stage(nxt, l0 + 32);
    wmma_tile_step<32, 128>(sKT[cur], sV[cur], wm, wn, lane, acc);
    async_wait_all();
    __syncthreads();
    cur = nxt;
  }

  const int hi = lane >> 4, ln = lane & 15;
  bf16* out = kvb + (size_t)bh * Dh * Dh;
#pragma unroll
  for (int mi = 0; mi < 4; ++mi)
#pragma unroll
    for (int ni = 0; ni < 2; ++ni)
#pragma unroll
      for (int r = 0; r < 8; ++r) {
        const int m = wm * 64 + mi * 16 + hi * 8 + r;
        const int n = wn * 32 + ni * 16 + ln;
        out[m * Dh + n] = (bf16)acc[mi][ni][r];
      }
}

// out = srms_norm(Q[b,h] @ kv[b,h]) * U, written bf16.
// Block = (b,h, 128-row tile); K = dh = 128 -> 4 WMMA k-steps.
// Q tile staged via async DMA; kv B-fragments come straight from global/L2.
__global__ __launch_bounds__(256)
void retention_norm_kernel(const bf16* __restrict__ Qb, const bf16* __restrict__ kvb,
                           const bf16* __restrict__ Ub, bf16* __restrict__ X2) {
  __shared__ __align__(16) bf16 sQ[128 * 128];
  __shared__ float rssq[128];
  const int bh = blockIdx.x, mt = blockIdx.y;
  const int b = bh >> 4, h = bh & 15;
  const int tid  = threadIdx.x;
  const int lane = tid & 31, wave = tid >> 5;
  const int wm = wave >> 2, wn = wave & 3;
  const size_t row0 = (size_t)b * Lsz + (size_t)mt * 128;

  { // stage the full 128x128 Q tile via async DMA (128B per thread)
    const int r = tid >> 1, c0 = (tid & 1) * 64;
    const bf16* src = Qb + (row0 + r) * Dsz + h * Dh + c0;
#pragma unroll
    for (int j = 0; j < 8; ++j)
      async_ld16(&sQ[r * 128 + c0 + j * 8], src + j * 8);
  }
  if (tid < 128) rssq[tid] = 0.0f;
  async_wait_all();
  __syncthreads();

  v8f acc[4][2] = {};
  const bf16* kvbase = kvb + (size_t)bh * Dh * Dh;
  const int kb    = (lane >> 4) * 8;
  const int am0   = wm * 64 + (lane & 15);
  const int bkrow = (lane & 15) + ((lane >> 4) << 4);
#pragma unroll
  for (int kk = 0; kk < 4; ++kk) {
    Frag a[4];
#pragma unroll
    for (int mi = 0; mi < 4; ++mi) {
      const bf16* p = sQ + (am0 + mi * 16) * 128 + kk * 32 + kb;
      a[mi].h[0] = *(const v8bf*)(p);
      a[mi].h[1] = *(const v8bf*)(p + 16);
    }
    Frag bfr[2];
#pragma unroll
    for (int ni = 0; ni < 2; ++ni) {
      const bf16* p = kvbase + (kk * 32) * Dh + (lane & 15) + wn * 32 + ni * 16;
#pragma unroll
      for (int s = 0; s < 8; ++s) {
        bfr[ni].v[s]     = p[(kb + s) * Dh];
        bfr[ni].v[8 + s] = p[(kb + 16 + s) * Dh];
      }
    }
#pragma unroll
    for (int mi = 0; mi < 4; ++mi)
#pragma unroll
      for (int ni = 0; ni < 2; ++ni)
        acc[mi][ni] = __builtin_amdgcn_wmma_f32_16x16x32_bf16(
            false, a[mi].v, false, bfr[ni].v, (short)0, acc[mi][ni], false, false);
  }

  // per-row sum of squares across the full head dim (128 cols span 4 wn-waves)
  const int hi = lane >> 4, ln = lane & 15;
#pragma unroll
  for (int mi = 0; mi < 4; ++mi)
#pragma unroll
    for (int r = 0; r < 8; ++r) {
      const int m = wm * 64 + mi * 16 + hi * 8 + r;
      const float p = acc[mi][0][r] * acc[mi][0][r] + acc[mi][1][r] * acc[mi][1][r];
      atomicAdd(&rssq[m], p);   // ds_add_f32
    }
  __syncthreads();

#pragma unroll
  for (int mi = 0; mi < 4; ++mi)
#pragma unroll
    for (int ni = 0; ni < 2; ++ni)
#pragma unroll
      for (int r = 0; r < 8; ++r) {
        const int m = wm * 64 + mi * 16 + hi * 8 + r;
        const int n = wn * 32 + ni * 16 + ln;
        const float norm = fmaxf(sqrtf(rssq[m] * (1.0f / 128.0f)), 1e-12f);
        const float v = acc[mi][ni][r] / norm;
        const size_t gidx = (row0 + m) * Dsz + h * Dh + n;
        const float u = (float)Ub[gidx];
        X2[gidx] = (bf16)(v * u);
      }
}

extern "C" void kernel_launch(void* const* d_in, const int* in_sizes, int n_in,
                              void* d_out, int out_size, void* d_ws, size_t ws_size,
                              hipStream_t stream) {
  (void)in_sizes; (void)n_in; (void)out_size; (void)ws_size;
  const float* x  = (const float*)d_in[0];
  const float* Wq = (const float*)d_in[1];
  const float* Wk = (const float*)d_in[2];
  const float* Wv = (const float*)d_in[3];
  const float* Wu = (const float*)d_in[4];
  const float* Wo = (const float*)d_in[5];

  // Workspace partition (bf16): Q,K,V,U,X2 (64 MiB each) + kv (2 MiB) = 322 MiB.
  const size_t mat = (size_t)Msz * Dsz;
  bf16* Qb  = (bf16*)d_ws;
  bf16* Kb  = Qb + mat;
  bf16* Vb  = Kb + mat;
  bf16* Ub  = Vb + mat;
  bf16* X2  = Ub + mat;
  bf16* kvb = X2 + mat;

  const dim3 blk(256);
  const dim3 g1(Dsz / 128, Msz / 128);  // 16 x 128 tiles

  gemm128_kernel<float, ACT_RELU_SCALE><<<g1, blk, 0, stream>>>(x, Wq, Qb, Msz, Dsz, Dsz);
  gemm128_kernel<float, ACT_RELU_SCALE><<<g1, blk, 0, stream>>>(x, Wk, Kb, Msz, Dsz, Dsz);
  gemm128_kernel<float, ACT_NONE_BF16 ><<<g1, blk, 0, stream>>>(x, Wv, Vb, Msz, Dsz, Dsz);
  gemm128_kernel<float, ACT_SILU      ><<<g1, blk, 0, stream>>>(x, Wu, Ub, Msz, Dsz, Dsz);

  kv_kernel<<<dim3(Bsz * Hn), blk, 0, stream>>>(Kb, Vb, kvb);
  retention_norm_kernel<<<dim3(Bsz * Hn, Lsz / 128), blk, 0, stream>>>(Qb, kvb, Ub, X2);

  gemm128_kernel<bf16, ACT_NONE_F32><<<g1, blk, 0, stream>>>(X2, Wo, (float*)d_out, Msz, Dsz, Dsz);
}
